// GAT_60756607369497
// MI455X (gfx1250) — matrix operand order repaired and checked by reading, not compiled, not run on hardware
//
#include <hip/hip_runtime.h>
#include <hip/hip_bf16.h>

// ---------------- problem constants ----------------
#define NCOMP   20000
#define NSEC    64
#define HID     128
#define INSZ    64
#define SEQT    32
#define EINTRA  640000
#define EINTER  4096
#define G3H     (3 * HID)     // 384
#define NT3H    (G3H / 16)    // 24 column tiles of gates

typedef __attribute__((ext_vector_type(16))) _Float16 v16h;
typedef __attribute__((ext_vector_type(8)))  _Float16 v8h;
typedef __attribute__((ext_vector_type(8)))  float    v8f;

// ---------------- WMMA helpers ----------------
__device__ __forceinline__ v8f wmma_f16(v16h a, v16h b, v8f c) {
  // D = A(16x32,f16) * B(32x16,f16) + C(16x16,f32)
  return __builtin_amdgcn_wmma_f32_16x16x32_f16(
      /*neg_a=*/false, a, /*neg_b=*/false, b,
      /*c_mod=*/(short)0, c, /*reuse_a=*/false, /*reuse_b=*/false);
}

// A-matrix 16x32 f16 fragment from row-major source (row stride in elements).
// This code's k order (matches B): lanes 0-15 -> rows M=0..15, K = 0..15;
//               lanes 16-31 -> same rows, K = 16..31.
__device__ __forceinline__ v16h load_a_frag(const _Float16* __restrict__ base,
                                            int stride, int k0, int lane) {
  const int r  = lane & 15;
  const int kh = (lane >> 4) << 4;            // 0 or 16
  const _Float16* p = base + (size_t)r * stride + k0 + kh;
  v8h lo = *(const v8h*)(p);                  // K = k0+kh .. +7
  v8h hi = *(const v8h*)(p + 8);              // K = k0+kh+8 .. +15
  v16h out;
#pragma unroll
  for (int i = 0; i < 8; ++i) { out[i] = lo[i]; out[i + 8] = hi[i]; }
  return out;
}

// B-matrix 32x16 f16 fragment computing X*W^T with W row-major [Nrows, K]:
// B(k, n) = W[n0+n, k0+k]. Lane n = lane&15; lane<16 holds K 0..15,
// lane>=16 holds K 16..31 -> 16 consecutive f16 = one 32B load.
__device__ __forceinline__ v16h load_b_frag(const _Float16* __restrict__ W,
                                            int ldw, int n0, int k0, int lane) {
  const int n  = n0 + (lane & 15);
  const int kh = (lane >> 4) << 4;            // 0 or 16
  return *(const v16h*)(W + (size_t)n * ldw + k0 + kh);
}

// monotonic float <-> u32 key (for atomic max on floats)
__device__ __forceinline__ unsigned fkey(float f) {
  unsigned u = __float_as_uint(f);
  return (u & 0x80000000u) ? ~u : (u | 0x80000000u);
}
__device__ __forceinline__ float fdecode(unsigned k) {
  unsigned u = (k & 0x80000000u) ? (k & 0x7FFFFFFFu) : ~k;
  return __uint_as_float(u);
}

// ---------------- conversion ----------------
__global__ void cvt_f32_to_f16_kernel(const float* __restrict__ in,
                                      _Float16* __restrict__ out, size_t n) {
  size_t i = (size_t)blockIdx.x * blockDim.x + threadIdx.x;
  if (i < n) out[i] = (_Float16)in[i];
}

// ---------------- fused GRU step (WMMA GEMM + cell) ----------------
// One block = 16 rows (companies). 8 waves; wave w computes gate column
// tiles 3w..3w+2 for BOTH gi = x_t*W_ih^T and gh = h*W_hh^T, stores the
// 16x384 tiles to LDS, then the block applies the GRU cell elementwise.
template <int K>
__global__ __launch_bounds__(256)
void gru_step_kernel(const _Float16* __restrict__ x, int xstride,
                     const _Float16* __restrict__ Wih,   // [3H, K] f16 row-major
                     const _Float16* __restrict__ Whh,   // [3H, H] f16 row-major
                     const float* __restrict__ bih, const float* __restrict__ bhh,
                     const _Float16* __restrict__ h_in_h, // [N, H] f16
                     const float*    __restrict__ h_in_f, // [N, H] f32
                     _Float16* __restrict__ h_out_h,
                     float*    __restrict__ h_out_f) {
  __shared__ float gi_s[16][G3H];
  __shared__ float gh_s[16][G3H];

  const int lane = threadIdx.x & 31;
  const int wave = threadIdx.x >> 5;
  const int row0 = blockIdx.x * 16;

  constexpr int KX = K / 32;
  const _Float16* xrow = x + (size_t)row0 * xstride;
  const _Float16* hrow = h_in_h + (size_t)row0 * HID;

  v16h ax[KX];
#pragma unroll
  for (int kt = 0; kt < KX; ++kt) ax[kt] = load_a_frag(xrow, xstride, kt * 32, lane);
  v16h ah[HID / 32];
#pragma unroll
  for (int kt = 0; kt < HID / 32; ++kt) ah[kt] = load_a_frag(hrow, HID, kt * 32, lane);

#pragma unroll
  for (int i = 0; i < 3; ++i) {
    const int nt = wave * 3 + i;
    const int n0 = nt * 16;
    v8f cgi = {};
#pragma unroll
    for (int kt = 0; kt < KX; ++kt)
      cgi = wmma_f16(ax[kt], load_b_frag(Wih, K, n0, kt * 32, lane), cgi);
    v8f cgh = {};
#pragma unroll
    for (int kt = 0; kt < HID / 32; ++kt)
      cgh = wmma_f16(ah[kt], load_b_frag(Whh, HID, n0, kt * 32, lane), cgh);

    // C layout: lane<16 -> N=lane, M=vgpr; lane>=16 -> N=lane-16, M=vgpr+8
    const int col = n0 + (lane & 15);
    const int mb  = (lane >> 4) << 3;
#pragma unroll
    for (int j = 0; j < 8; ++j) {
      gi_s[mb + j][col] = cgi[j];
      gh_s[mb + j][col] = cgh[j];
    }
  }
  __syncthreads();

  // GRU cell over 16 x 128 elements
  for (int idx = threadIdx.x; idx < 16 * HID; idx += 256) {
    const int m = idx >> 7;
    const int c = idx & (HID - 1);
    const float ir = gi_s[m][c]           + bih[c];
    const float iz = gi_s[m][c + HID]     + bih[c + HID];
    const float in_ = gi_s[m][c + 2*HID]  + bih[c + 2*HID];
    const float hr = gh_s[m][c]           + bhh[c];
    const float hz = gh_s[m][c + HID]     + bhh[c + HID];
    const float hn = gh_s[m][c + 2*HID]   + bhh[c + 2*HID];
    const float r = 1.0f / (1.0f + __expf(-(ir + hr)));
    const float z = 1.0f / (1.0f + __expf(-(iz + hz)));
    const float nn = tanhf(in_ + r * hn);
    const size_t g = (size_t)(row0 + m) * HID + c;
    const float hnew = (1.0f - z) * nn + z * h_in_f[g];
    h_out_f[g] = hnew;
    h_out_h[g] = (_Float16)hnew;
  }
}

// ---------------- generic C[M,128] = A[M,128] * W[128,128]^T ----------------
__global__ __launch_bounds__(256)
void gemm128_kernel(const _Float16* __restrict__ A,
                    const _Float16* __restrict__ W,
                    float* __restrict__ C, int M) {
  const int lane = threadIdx.x & 31;
  const int wave = threadIdx.x >> 5;
  const int row0 = blockIdx.x * 16;
  if (row0 >= M) return;

  v16h a[HID / 32];
#pragma unroll
  for (int kt = 0; kt < HID / 32; ++kt)
    a[kt] = load_a_frag(A + (size_t)row0 * HID, HID, kt * 32, lane);

  const int n0 = wave * 16;                 // 8 waves * 16 = 128 cols
  v8f c = {};
#pragma unroll
  for (int kt = 0; kt < HID / 32; ++kt)
    c = wmma_f16(a[kt], load_b_frag(W, HID, n0, kt * 32, lane), c);

  const int col = n0 + (lane & 15);
  const int mb  = (lane >> 4) << 3;
#pragma unroll
  for (int j = 0; j < 8; ++j)
    C[(size_t)(row0 + mb + j) * HID + col] = c[j];
}

// ---------------- GAT attention scores: e = h . a ----------------
__global__ void att_scores_kernel(const float* __restrict__ h,
                                  const float* __restrict__ a_src,
                                  const float* __restrict__ a_dst,
                                  float* __restrict__ esrc, float* __restrict__ edst,
                                  int M) {
  const int gid  = blockIdx.x * blockDim.x + threadIdx.x;
  const int wid  = gid >> 5;
  const int lane = gid & 31;
  if (wid >= M) return;
  float s1 = 0.0f, s2 = 0.0f;
  for (int c = lane; c < HID; c += 32) {
    const float v = h[(size_t)wid * HID + c];
    s1 += v * a_src[c];
    s2 += v * a_dst[c];
  }
#pragma unroll
  for (int off = 16; off > 0; off >>= 1) {
    s1 += __shfl_xor(s1, off, 32);
    s2 += __shfl_xor(s2, off, 32);
  }
  if (lane == 0) { esrc[wid] = s1; edst[wid] = s2; }
}

// ---------------- edge softmax: pass 1 (segment max via key trick) ----------
__global__ void edge_max_kernel(const int* __restrict__ src, const int* __restrict__ dst,
                                int E, int Nn,
                                const float* __restrict__ esrc, const float* __restrict__ edst,
                                unsigned* __restrict__ keys) {
  const int e = blockIdx.x * blockDim.x + threadIdx.x;
  if (e >= E + Nn) return;
  int s, d;
  if (e < E) { s = src[e]; d = dst[e]; } else { s = d = e - E; }  // self loop
  float v = esrc[s] + edst[d];
  v = (v > 0.0f) ? v : 0.2f * v;                                  // leaky relu
  atomicMax(&keys[d], fkey(v));
}

// ---------------- edge softmax: pass 2 (weighted accumulate) ----------------
__global__ void edge_accum_kernel(const int* __restrict__ src, const int* __restrict__ dst,
                                  int E, int Nn,
                                  const float* __restrict__ esrc, const float* __restrict__ edst,
                                  const unsigned* __restrict__ keys,
                                  const float* __restrict__ h,
                                  float* __restrict__ acc, float* __restrict__ denom) {
  const int gid  = blockIdx.x * blockDim.x + threadIdx.x;
  const int e    = gid >> 5;
  const int lane = gid & 31;
  if (e >= E + Nn) return;
  int s, d;
  if (e < E) { s = src[e]; d = dst[e]; } else { s = d = e - E; }
  float v = esrc[s] + edst[d];
  v = (v > 0.0f) ? v : 0.2f * v;
  const float w = __expf(v - fdecode(keys[d]));
  if (lane == 0) atomicAdd(&denom[d], w);
  for (int c = lane; c < HID; c += 32)
    atomicAdd(&acc[(size_t)d * HID + c], w * h[(size_t)s * HID + c]);
}

// ---------------- finalize: out = acc/denom + bias ----------------
__global__ void gat_finalize_kernel(const float* __restrict__ acc,
                                    const float* __restrict__ denom,
                                    const float* __restrict__ bias,
                                    float* __restrict__ out, int Nn) {
  const int idx = blockIdx.x * blockDim.x + threadIdx.x;
  if (idx >= Nn * HID) return;
  const int n = idx >> 7;
  const int c = idx & (HID - 1);
  out[idx] = acc[idx] / denom[n] + bias[c];
}

// ---------------- per-sector max pool ----------------
__global__ void pool_max_kernel(const float* __restrict__ x,
                                const int* __restrict__ sector,
                                unsigned* __restrict__ keys, int Nn) {
  const int idx = blockIdx.x * blockDim.x + threadIdx.x;
  if (idx >= Nn * HID) return;
  const int n = idx >> 7;
  const int c = idx & (HID - 1);
  atomicMax(&keys[sector[n] * HID + c], fkey(x[idx]));
}

__global__ void key_decode_kernel(const unsigned* __restrict__ keys,
                                  float* __restrict__ outf,
                                  _Float16* __restrict__ outh, int count) {
  const int i = blockIdx.x * blockDim.x + threadIdx.x;
  if (i >= count) return;
  const float v = fdecode(keys[i]);
  outf[i] = v;
  outh[i] = (_Float16)v;
}

// ---------------- fusion head ----------------
__global__ void fusion_kernel(const float* __restrict__ seq,
                              const float* __restrict__ intra,
                              const float* __restrict__ inter_sec,
                              const int* __restrict__ sector,
                              const float* __restrict__ fW, const float* __restrict__ fb,
                              float* __restrict__ out, int Nn) {
  const int n = blockIdx.x * blockDim.x + threadIdx.x;
  if (n >= Nn) return;
  const int sct = sector[n];
  float acc = fb[0];
  for (int c = 0; c < HID; ++c) {
    acc += seq[(size_t)n * HID + c]       * fW[c];
    acc += intra[(size_t)n * HID + c]     * fW[HID + c];
    acc += inter_sec[(size_t)sct * HID + c] * fW[2 * HID + c];
  }
  out[n] = acc;
}

// ---------------- host ----------------
static inline void* ws_alloc(char*& p, size_t bytes) {
  void* r = (void*)p;
  p += (bytes + 255) & ~(size_t)255;
  return r;
}

extern "C" void kernel_launch(void* const* d_in, const int* in_sizes, int n_in,
                              void* d_out, int out_size, void* d_ws, size_t ws_size,
                              hipStream_t stream) {
  (void)in_sizes; (void)n_in; (void)out_size; (void)ws_size;

  const float* x        = (const float*)d_in[0];
  const int*   intra_ei = (const int*)d_in[1];
  const int*   inter_ei = (const int*)d_in[2];
  const int*   sector   = (const int*)d_in[3];
  const float* wih0 = (const float*)d_in[4];
  const float* whh0 = (const float*)d_in[5];
  const float* bih0 = (const float*)d_in[6];
  const float* bhh0 = (const float*)d_in[7];
  const float* wih1 = (const float*)d_in[8];
  const float* whh1 = (const float*)d_in[9];
  const float* bih1 = (const float*)d_in[10];
  const float* bhh1 = (const float*)d_in[11];
  const float* intraW   = (const float*)d_in[12];
  const float* intraAs  = (const float*)d_in[13];
  const float* intraAd  = (const float*)d_in[14];
  const float* intraB   = (const float*)d_in[15];
  const float* interW   = (const float*)d_in[16];
  const float* interAs  = (const float*)d_in[17];
  const float* interAd  = (const float*)d_in[18];
  const float* interB   = (const float*)d_in[19];
  const float* fusW     = (const float*)d_in[20];
  const float* fusB     = (const float*)d_in[21];
  float* out = (float*)d_out;

  // ---- workspace layout ----
  char* p = (char*)d_ws;
  _Float16* xh     = (_Float16*)ws_alloc(p, (size_t)NCOMP * SEQT * INSZ * 2);
  _Float16* Wih0h  = (_Float16*)ws_alloc(p, (size_t)G3H * INSZ * 2);
  _Float16* Whh0h  = (_Float16*)ws_alloc(p, (size_t)G3H * HID * 2);
  _Float16* Wih1h  = (_Float16*)ws_alloc(p, (size_t)G3H * HID * 2);
  _Float16* Whh1h  = (_Float16*)ws_alloc(p, (size_t)G3H * HID * 2);
  _Float16* intraWh = (_Float16*)ws_alloc(p, (size_t)HID * HID * 2);
  _Float16* interWh = (_Float16*)ws_alloc(p, (size_t)HID * HID * 2);
  _Float16* h0h[2]; float* h0f[2]; _Float16* h1h[2]; float* h1f[2];
  for (int i = 0; i < 2; ++i) {
    h0h[i] = (_Float16*)ws_alloc(p, (size_t)NCOMP * HID * 2);
    h0f[i] = (float*)   ws_alloc(p, (size_t)NCOMP * HID * 4);
    h1h[i] = (_Float16*)ws_alloc(p, (size_t)NCOMP * HID * 2);
    h1f[i] = (float*)   ws_alloc(p, (size_t)NCOMP * HID * 4);
  }
  float*    h_intra  = (float*)   ws_alloc(p, (size_t)NCOMP * HID * 4);
  float*    esrcN    = (float*)   ws_alloc(p, (size_t)NCOMP * 4);
  float*    edstN    = (float*)   ws_alloc(p, (size_t)NCOMP * 4);
  unsigned* keysN    = (unsigned*)ws_alloc(p, (size_t)NCOMP * 4);
  float*    denomN   = (float*)   ws_alloc(p, (size_t)NCOMP * 4);
  float*    accN     = (float*)   ws_alloc(p, (size_t)NCOMP * HID * 4);
  float*    intraEmb = (float*)   ws_alloc(p, (size_t)NCOMP * HID * 4);
  unsigned* poolKeys = (unsigned*)ws_alloc(p, (size_t)NSEC * HID * 4);
  float*    poolF    = (float*)   ws_alloc(p, (size_t)NSEC * HID * 4);
  _Float16* poolH    = (_Float16*)ws_alloc(p, (size_t)NSEC * HID * 2);
  float*    h_inter  = (float*)   ws_alloc(p, (size_t)NSEC * HID * 4);
  float*    esrcS    = (float*)   ws_alloc(p, (size_t)NSEC * 4);
  float*    edstS    = (float*)   ws_alloc(p, (size_t)NSEC * 4);
  unsigned* keysS    = (unsigned*)ws_alloc(p, (size_t)NSEC * 4);
  float*    denomS   = (float*)   ws_alloc(p, (size_t)NSEC * 4);
  float*    accS     = (float*)   ws_alloc(p, (size_t)NSEC * HID * 4);
  float*    interSec = (float*)   ws_alloc(p, (size_t)NSEC * HID * 4);

  // ---- f32 -> f16 conversions ----
  {
    size_t nx = (size_t)NCOMP * SEQT * INSZ;
    cvt_f32_to_f16_kernel<<<(unsigned)((nx + 255) / 256), 256, 0, stream>>>(x, xh, nx);
    cvt_f32_to_f16_kernel<<<(G3H * INSZ + 255) / 256, 256, 0, stream>>>(wih0, Wih0h, (size_t)G3H * INSZ);
    cvt_f32_to_f16_kernel<<<(G3H * HID + 255) / 256, 256, 0, stream>>>(whh0, Whh0h, (size_t)G3H * HID);
    cvt_f32_to_f16_kernel<<<(G3H * HID + 255) / 256, 256, 0, stream>>>(wih1, Wih1h, (size_t)G3H * HID);
    cvt_f32_to_f16_kernel<<<(G3H * HID + 255) / 256, 256, 0, stream>>>(whh1, Whh1h, (size_t)G3H * HID);
    cvt_f32_to_f16_kernel<<<(HID * HID + 255) / 256, 256, 0, stream>>>(intraW, intraWh, (size_t)HID * HID);
    cvt_f32_to_f16_kernel<<<(HID * HID + 255) / 256, 256, 0, stream>>>(interW, interWh, (size_t)HID * HID);
  }

  // ---- zero initial hidden state (buffer 0 = input at t=0) ----
  hipMemsetAsync(h0h[0], 0, (size_t)NCOMP * HID * 2, stream);
  hipMemsetAsync(h0f[0], 0, (size_t)NCOMP * HID * 4, stream);
  hipMemsetAsync(h1h[0], 0, (size_t)NCOMP * HID * 2, stream);
  hipMemsetAsync(h1f[0], 0, (size_t)NCOMP * HID * 4, stream);

  // ---- 2-layer GRU, interleaved timesteps (out0 never materialized) ----
  const int nRowBlk = NCOMP / 16;   // 1250
  for (int t = 0; t < SEQT; ++t) {
    const int a = t & 1, b = (t + 1) & 1;
    gru_step_kernel<INSZ><<<nRowBlk, 256, 0, stream>>>(
        xh + (size_t)t * INSZ, SEQT * INSZ, Wih0h, Whh0h, bih0, bhh0,
        h0h[a], h0f[a], h0h[b], h0f[b]);
    gru_step_kernel<HID><<<nRowBlk, 256, 0, stream>>>(
        h0h[b], HID, Wih1h, Whh1h, bih1, bhh1,
        h1h[a], h1f[a], h1h[b], h1f[b]);
  }
  const _Float16* seqH = h1h[0];   // (31+1)&1 == 0
  const float*    seqF = h1f[0];

  // ---- intra GAT ----
  gemm128_kernel<<<nRowBlk, 256, 0, stream>>>(seqH, intraWh, h_intra, NCOMP);
  att_scores_kernel<<<(NCOMP + 7) / 8, 256, 0, stream>>>(h_intra, intraAs, intraAd, esrcN, edstN, NCOMP);
  hipMemsetAsync(keysN, 0, (size_t)NCOMP * 4, stream);
  hipMemsetAsync(denomN, 0, (size_t)NCOMP * 4, stream);
  hipMemsetAsync(accN, 0, (size_t)NCOMP * HID * 4, stream);
  {
    const int edges = EINTRA + NCOMP;
    edge_max_kernel<<<(edges + 255) / 256, 256, 0, stream>>>(
        intra_ei, intra_ei + EINTRA, EINTRA, NCOMP, esrcN, edstN, keysN);
    edge_accum_kernel<<<(edges * 32 + 255) / 256, 256, 0, stream>>>(
        intra_ei, intra_ei + EINTRA, EINTRA, NCOMP, esrcN, edstN, keysN,
        h_intra, accN, denomN);
  }
  gat_finalize_kernel<<<(NCOMP * HID + 255) / 256, 256, 0, stream>>>(accN, denomN, intraB, intraEmb, NCOMP);

  // ---- per-sector max pool ----
  hipMemsetAsync(poolKeys, 0, (size_t)NSEC * HID * 4, stream);
  pool_max_kernel<<<(NCOMP * HID + 255) / 256, 256, 0, stream>>>(intraEmb, sector, poolKeys, NCOMP);
  key_decode_kernel<<<(NSEC * HID + 255) / 256, 256, 0, stream>>>(poolKeys, poolF, poolH, NSEC * HID);

  // ---- inter GAT (64 sector nodes) ----
  gemm128_kernel<<<NSEC / 16, 256, 0, stream>>>(poolH, interWh, h_inter, NSEC);
  att_scores_kernel<<<(NSEC + 7) / 8, 256, 0, stream>>>(h_inter, interAs, interAd, esrcS, edstS, NSEC);
  hipMemsetAsync(keysS, 0, (size_t)NSEC * 4, stream);
  hipMemsetAsync(denomS, 0, (size_t)NSEC * 4, stream);
  hipMemsetAsync(accS, 0, (size_t)NSEC * HID * 4, stream);
  {
    const int edges = EINTER + NSEC;
    edge_max_kernel<<<(edges + 255) / 256, 256, 0, stream>>>(
        inter_ei, inter_ei + EINTER, EINTER, NSEC, esrcS, edstS, keysS);
    edge_accum_kernel<<<(edges * 32 + 255) / 256, 256, 0, stream>>>(
        inter_ei, inter_ei + EINTER, EINTER, NSEC, esrcS, edstS, keysS,
        h_inter, accS, denomS);
  }
  gat_finalize_kernel<<<(NSEC * HID + 255) / 256, 256, 0, stream>>>(accS, denomS, interB, interSec, NSEC);

  // ---- fusion head ----
  fusion_kernel<<<(NCOMP + 255) / 256, 256, 0, stream>>>(
      seqF, intraEmb, interSec, sector, fusW, fusB, out, NCOMP);
}
